// ProxyNCALayer_75359496176232
// MI455X (gfx1250) — compile-verified, matrix operand order checked by reading, not run on hardware
//
#include <hip/hip_runtime.h>

// ProxyNCA distance: out[b,c] = ||x_b||^2 + ||p_c||^2 - 2 x_b . p_c
// x = 3*l2norm(inputs), p = 3*l2norm(kernel).  B=4096, C=16384, D=128 (fp32).
//
// MI455X reasoning: output is 256MB fp32 streamed to HBM -> store-bound
// (~11us at 23.3 TB/s). Cross-term GEMM (17.2 GFLOP) runs on
// v_wmma_f32_16x16x32_f16 with fp32 accumulation; normalized values are
// bounded (<=3) so f16 operands are accurate. Norms added in fp32 epilogue.

typedef __attribute__((ext_vector_type(4)))  _Float16 v4h;
typedef __attribute__((ext_vector_type(8)))  _Float16 v8h;
typedef __attribute__((ext_vector_type(16))) _Float16 v16h;
typedef __attribute__((ext_vector_type(8)))  float    v8f;

#define DDIM 128

// ---------------- row normalization: one wave32 per row ----------------
__global__ __launch_bounds__(256)
void norm_rows_kernel(const float* __restrict__ src,
                      _Float16* __restrict__ dsth,
                      float* __restrict__ nrm, int nrows)
{
    int row  = blockIdx.x * 8 + (threadIdx.x >> 5);
    int lane = threadIdx.x & 31;
    if (row >= nrows) return;

    float4 v = ((const float4*)src)[(size_t)row * (DDIM / 4) + lane];
    float s = v.x * v.x + v.y * v.y + v.z * v.z + v.w * v.w;
#pragma unroll
    for (int o = 16; o > 0; o >>= 1) s += __shfl_xor(s, o);

    float scale = 3.0f * rsqrtf(fmaxf(s, 1e-12f));   // 3 / sqrt(max(sq,eps))
    v4h h;
    h[0] = (_Float16)(v.x * scale);
    h[1] = (_Float16)(v.y * scale);
    h[2] = (_Float16)(v.z * scale);
    h[3] = (_Float16)(v.w * scale);
    *(v4h*)(dsth + (size_t)row * DDIM + lane * 4) = h;
    if (lane == 0) nrm[row] = s * scale * scale;     // sum of normalized^2
}

// ---------------- WMMA GEMM + epilogue ----------------
__device__ __forceinline__ v16h cat8(v8h lo, v8h hi)
{
    v16h r;
#pragma unroll
    for (int i = 0; i < 8; ++i) { r[i] = lo[i]; r[i + 8] = hi[i]; }
    return r;
}

// Fragment load from LDS (xor-swizzled 16B chunks) matching the ISA's
// 16-bit 16x32 A layout (B uses the mirrored per-column layout -> same code):
// lanes 0-15 hold K=[k0..k0+7] and [k0+16..k0+23]; lanes 16-31 the +8 halves.
__device__ __forceinline__ v16h frag_ld(const _Float16* s, int row0, int k0, int lane)
{
    int lh = lane >> 4;                    // lane half selects +8 K offset
    int r  = row0 + (lane & 15);
    int c0 = (k0 >> 3) + lh;               // 16B-chunk index of first half
    int sw = r & 15;                       // xor swizzle key
    const _Float16* rp = s + r * DDIM;
    v8h lo = *(const v8h*)(rp + ((c0    ) ^ sw) * 8);
    v8h hi = *(const v8h*)(rp + ((c0 + 2) ^ sw) * 8);
    return cat8(lo, hi);
}

__global__ __launch_bounds__(256)
void pnca_gemm_kernel(const _Float16* __restrict__ xh,
                      const _Float16* __restrict__ ph,
                      const float* __restrict__ xs,
                      const float* __restrict__ ps,
                      float* __restrict__ out, int Ccols)
{
    __shared__ alignas(16) _Float16 sA[128 * DDIM];   // 32 KB
    __shared__ alignas(16) _Float16 sB[128 * DDIM];   // 32 KB

    int t     = threadIdx.x;
    int mBase = blockIdx.y * 128;
    int nBase = blockIdx.x * 128;

    // Stage 128x128-half tiles of A (x) and B (p) into LDS.
    // 2048 16B chunks per tile, 8 per thread, xor-swizzled by row.
    const uint4* gA = (const uint4*)xh + (size_t)mBase * (DDIM / 8);
    const uint4* gB = (const uint4*)ph + (size_t)nBase * (DDIM / 8);
#pragma unroll
    for (int i = 0; i < 8; ++i) {
        int id = t + i * 256;              // 0..2047
        int r  = id >> 4;
        int c  = id & 15;
        int cs = c ^ (r & 15);
        ((uint4*)sA)[r * 16 + cs] = gA[r * 16 + c];
        ((uint4*)sB)[r * 16 + cs] = gB[r * 16 + c];
    }
    __syncthreads();

    int w    = t >> 5;
    int lane = t & 31;
    int m0w  = (w >> 1) * 32;              // 4 wave-rows  x 32
    int n0w  = (w & 1) * 64;               // 2 wave-cols  x 64

    v8f acc[2][4] = {};
#pragma unroll
    for (int ks = 0; ks < 4; ++ks) {       // K = 128 in 4 steps of 32
        int k0 = ks * 32;
        v16h a[2], b[4];
#pragma unroll
        for (int mt = 0; mt < 2; ++mt) a[mt] = frag_ld(sA, m0w + mt * 16, k0, lane);
#pragma unroll
        for (int nt = 0; nt < 4; ++nt) b[nt] = frag_ld(sB, n0w + nt * 16, k0, lane);
#pragma unroll
        for (int mt = 0; mt < 2; ++mt)
#pragma unroll
            for (int nt = 0; nt < 4; ++nt)
                acc[mt][nt] = __builtin_amdgcn_wmma_f32_16x16x32_f16(
                    false, a[mt], false, b[nt], (short)0, acc[mt][nt],
                    false, false);
    }

    // Epilogue: out = xs[row] + ps[col] - 2*cross, streamed with NT stores.
    int lh = lane >> 4;
    int ln = lane & 15;
    float psv[4], xsv[2][8];
#pragma unroll
    for (int nt = 0; nt < 4; ++nt) psv[nt] = ps[nBase + n0w + nt * 16 + ln];
#pragma unroll
    for (int mt = 0; mt < 2; ++mt)
#pragma unroll
        for (int j = 0; j < 8; ++j)
            xsv[mt][j] = xs[mBase + m0w + mt * 16 + lh * 8 + j];

#pragma unroll
    for (int mt = 0; mt < 2; ++mt)
#pragma unroll
        for (int nt = 0; nt < 4; ++nt) {
#pragma unroll
            for (int j = 0; j < 8; ++j) {   // VGPR j: M=j (lanes 0-15), M=j+8 (16-31)
                int row = mBase + m0w + mt * 16 + lh * 8 + j;
                int col = nBase + n0w + nt * 16 + ln;
                float o = xsv[mt][j] + psv[nt] - 2.0f * acc[mt][nt][j];
                __builtin_nontemporal_store(o, out + (size_t)row * Ccols + col);
            }
        }
}

// ---------------- host-side launch ----------------
extern "C" void kernel_launch(void* const* d_in, const int* in_sizes, int n_in,
                              void* d_out, int out_size, void* d_ws, size_t ws_size,
                              hipStream_t stream)
{
    const float* inp = (const float*)d_in[0];   // [B, 128] fp32
    const float* ker = (const float*)d_in[1];   // [C, 128] fp32
    float* out = (float*)d_out;                 // [B, C] fp32

    int B = in_sizes[0] / DDIM;                 // 4096
    int C = in_sizes[1] / DDIM;                 // 16384

    // Workspace layout: xh[B*128] f16 | ph[C*128] f16 | xs[B] f32 | ps[C] f32
    char* ws = (char*)d_ws;
    _Float16* xh = (_Float16*)ws;
    _Float16* ph = (_Float16*)(ws + (size_t)B * DDIM * sizeof(_Float16));
    float*    xs = (float*)(ws + (size_t)(B + C) * DDIM * sizeof(_Float16));
    float*    ps = xs + B;

    norm_rows_kernel<<<dim3((B + 7) / 8), 256, 0, stream>>>(inp, xh, xs, B);
    norm_rows_kernel<<<dim3((C + 7) / 8), 256, 0, stream>>>(ker, ph, ps, C);
    pnca_gemm_kernel<<<dim3(C / 128, B / 128), 256, 0, stream>>>(xh, ph, xs, ps, out, C);
}
